// CompilablesLSTMBlock_70428873719998
// MI455X (gfx1250) — matrix operand order repaired and checked by reading, not compiled, not run on hardware
//
#include <hip/hip_runtime.h>
#include <hip/hip_bf16.h>

// ---------------------------------------------------------------------------
// sLSTM block for MI455X (gfx1250, wave32, WMMA).
//   B=4, S=4096, D=1024, P=1364, 3P=4092, fused gate+cell N = 4P = 5456
//   N padded to NSTRIDE=5504 (= 86 * 64): every wave computes a full 32x64
//   tile with EXEC all-ones (WMMA requirement), zero divergent guards.
// Pipeline:
//   1) RMSNorm -> xn (bf16)                       [33MB, L2 resident]
//   2) transpose-convert W_gate|W_cell -> WT bf16 [5504,1024] (pad rows = 0)
//      transpose-convert W_out -> WoT bf16 [1024,1376] (K padded)
//   3) WMMA GEMM: G[16384,5504] = xn @ WT^T   (fp32, NT stores)
//   4) chunked associative scan (64 chunks x 64 steps)
//   5) WMMA GEMM + residual: out = x + outputs @ W_out
//   6) copy initial hidden_state to tail of d_out
// ---------------------------------------------------------------------------

#define BB      4
#define SS      4096
#define DD      1024
#define PP      1364
#define NCOLS   (4 * PP)        // 5456 valid fused columns
#define NSTRIDE 5504            // padded to 86 * 64
#define NJOBS1  (NSTRIDE / 64)  // 86
#define MM      (BB * SS)       // 16384 rows
#define KPAD    1376            // P padded to multiple of 32 for out-GEMM
#define LCH     64              // steps per chunk
#define NCHUNK  (SS / LCH)      // 64 chunks
#define BP      (BB * PP)       // 5456 channels
#define CAPF    15.0f

typedef __attribute__((ext_vector_type(16))) __bf16 bf16x16;
typedef __attribute__((ext_vector_type(8)))  float  f32x8;

struct U4x2 { uint4 lo, hi; };

// ---- WMMA fragment loaders (row-major global_load_b128 per ISA layout) ----
__device__ __forceinline__ bf16x16 load_a_frag(const __bf16* __restrict__ A,
                                               int lda, int m0, int kk, int lane) {
  // 16-bit A 16x32: lanes 0-15 -> M=lane, K in {0..7,16..23}; lanes 16-31 -> K in {8..15,24..31}
  const int row  = m0 + (lane & 15);
  const int half = lane >> 4;
  const __bf16* p = A + (size_t)row * lda + kk + half * 8;
  U4x2 t;
  t.lo = *(const uint4*)(p);        // K: half*8 .. half*8+7
  t.hi = *(const uint4*)(p + 16);   // K: 16+half*8 .. 16+half*8+7
  return __builtin_bit_cast(bf16x16, t);
}

__device__ __forceinline__ bf16x16 load_b_frag(const __bf16* __restrict__ BT,
                                               int ldb, int n0, int kk, int lane) {
  // 16-bit B 32x16 from transposed weights (row-major [N,K]):
  // lanes 0-15 -> N=lane, K in {0..7,16..23}; lanes 16-31 -> K in {8..15,24..31}
  const int col  = n0 + (lane & 15);
  const int half = lane >> 4;
  const __bf16* p = BT + (size_t)col * ldb + kk + half * 8;
  U4x2 t;
  t.lo = *(const uint4*)(p);
  t.hi = *(const uint4*)(p + 16);
  return __builtin_bit_cast(bf16x16, t);
}

// ---------------------------------------------------------------------------
// 1) RMSNorm -> bf16
// ---------------------------------------------------------------------------
__global__ void k_rmsnorm_bf16(const float* __restrict__ X,
                               const float* __restrict__ w,
                               __bf16* __restrict__ Xn) {
  const int row = blockIdx.x;                       // 0..MM-1
  const float* xr = X + (size_t)row * DD;
  float s = 0.f;
  for (int i = threadIdx.x; i < DD; i += 256) { float v = xr[i]; s += v * v; }
  __shared__ float red[256];
  red[threadIdx.x] = s; __syncthreads();
  for (int off = 128; off > 0; off >>= 1) {
    if (threadIdx.x < off) red[threadIdx.x] += red[threadIdx.x + off];
    __syncthreads();
  }
  const float rinv = rsqrtf(red[0] * (1.0f / (float)DD) + 1e-6f);
  for (int i = threadIdx.x; i < DD; i += 256)
    Xn[(size_t)row * DD + i] = (__bf16)(xr[i] * rinv * w[i]);
}

// ---------------------------------------------------------------------------
// 2) Weight transpose-convert (LDS tiled, coalesced both sides)
// ---------------------------------------------------------------------------
__global__ void k_build_wt_fused(const float* __restrict__ Wg,   // [D, 3P]
                                 const float* __restrict__ Wc,   // [D, P]
                                 __bf16* __restrict__ WT) {      // [NSTRIDE, D]
  __shared__ float tile[32][33];
  const int k0 = blockIdx.x * 32;   // over D (1024, exact)
  const int n0 = blockIdx.y * 32;   // over NSTRIDE (5504, exact)
  const int tx = threadIdx.x, ty = threadIdx.y;
#pragma unroll
  for (int i = 0; i < 4; ++i) {
    const int k = k0 + ty + i * 8, n = n0 + tx;
    float v = 0.f;
    if (n < 3 * PP)      v = Wg[(size_t)k * (3 * PP) + n];
    else if (n < NCOLS)  v = Wc[(size_t)k * PP + (n - 3 * PP)];
    tile[ty + i * 8][tx] = v;       // pad rows (n >= NCOLS) -> 0
  }
  __syncthreads();
#pragma unroll
  for (int i = 0; i < 4; ++i) {
    const int n = n0 + ty + i * 8, k = k0 + tx;
    WT[(size_t)n * DD + k] = (__bf16)tile[tx][ty + i * 8];
  }
}

__global__ void k_build_wout_t(const float* __restrict__ Wo,   // [P, D]
                               __bf16* __restrict__ WoT) {     // [D, KPAD]
  __shared__ float tile[32][33];
  const int p0 = blockIdx.x * 32;   // over KPAD (1376, exact)
  const int n0 = blockIdx.y * 32;   // over D (exact)
  const int tx = threadIdx.x, ty = threadIdx.y;
#pragma unroll
  for (int i = 0; i < 4; ++i) {
    const int p = p0 + ty + i * 8, n = n0 + tx;
    tile[ty + i * 8][tx] = (p < PP) ? Wo[(size_t)p * DD + n] : 0.f;
  }
  __syncthreads();
#pragma unroll
  for (int i = 0; i < 4; ++i) {
    const int n = n0 + ty + i * 8, p = p0 + tx;
    WoT[(size_t)n * KPAD + p] = (__bf16)tile[tx][ty + i * 8];
  }
}

// ---------------------------------------------------------------------------
// 3) Fused gate+cell GEMM: G[M, NSTRIDE] = Xn[M,D] @ WT^T ; fp32 NT stores.
//    Wave tile 32x64 (2 M x 4 N sub-tiles), K-step 32, guard-free.
//    grid = (NJOBS1, MM/256), block = 256 (8 waves -> 8 consecutive 32-row
//    M groups). Every B fragment feeds two WMMAs (1.5 loads/WMMA).
// ---------------------------------------------------------------------------
__global__ void k_gemm_gates(const __bf16* __restrict__ Xn,
                             const __bf16* __restrict__ WT,
                             float* __restrict__ G) {
  const int lane = threadIdx.x & 31;
  const int wave = threadIdx.x >> 5;
  const int m0   = (blockIdx.y * 8 + wave) * 32;
  const int n0   = blockIdx.x * 64;
  f32x8 acc0[4] = {}, acc1[4] = {};
  for (int kk = 0; kk < DD; kk += 32) {
    const bf16x16 a0 = load_a_frag(Xn, DD, m0,      kk, lane);
    const bf16x16 a1 = load_a_frag(Xn, DD, m0 + 16, kk, lane);
#pragma unroll
    for (int t = 0; t < 4; ++t) {
      const bf16x16 b = load_b_frag(WT, DD, n0 + t * 16, kk, lane);
      acc0[t] = __builtin_amdgcn_wmma_f32_16x16x32_bf16(
          false, a0, false, b, (short)0, acc0[t], false, false);
      acc1[t] = __builtin_amdgcn_wmma_f32_16x16x32_bf16(
          false, a1, false, b, (short)0, acc1[t], false, false);
    }
  }
  const int mb0 = m0 + (lane >> 4) * 8;
#pragma unroll
  for (int t = 0; t < 4; ++t) {
    const int n = n0 + t * 16 + (lane & 15);
#pragma unroll
    for (int r = 0; r < 8; ++r) {
      __builtin_nontemporal_store(acc0[t][r],
                                  &G[(size_t)(mb0 + r) * NSTRIDE + n]);
      __builtin_nontemporal_store(acc1[t][r],
                                  &G[(size_t)(mb0 + 16 + r) * NSTRIDE + n]);
    }
  }
}

// ---------------------------------------------------------------------------
// 4) Chunked associative scan:  h_t = f_t h + i_t tanh(c_t)
// ---------------------------------------------------------------------------
__device__ __forceinline__ float softcap_sig(float g) {
  const float t = CAPF * tanhf(g * (1.0f / CAPF));
  return 1.0f / (1.0f + __expf(-t));
}

__global__ void k_scan_chunks(const float* __restrict__ G,
                              float* __restrict__ Fc, float* __restrict__ Uc) {
  const int c = blockIdx.x * 256 + threadIdx.x;
  if (c >= BP) return;
  const int j = blockIdx.y;
  const int b = c / PP, p = c % PP;
  float F = 1.f, U = 0.f;
  const float* base = G + ((size_t)b * SS + (size_t)j * LCH) * NSTRIDE;
  for (int t = 0; t < LCH; ++t) {
    const float* row = base + (size_t)t * NSTRIDE;
    const float gi = __builtin_nontemporal_load(row + p);
    const float gf = __builtin_nontemporal_load(row + PP + p);
    const float gc = __builtin_nontemporal_load(row + 3 * PP + p);
    const float ff = softcap_sig(gf);
    F = F * ff;
    U = ff * U + softcap_sig(gi) * tanhf(gc);
  }
  Fc[(size_t)j * BP + c] = F;
  Uc[(size_t)j * BP + c] = U;
}

__global__ void k_scan_prefix(const float* __restrict__ Fc,
                              const float* __restrict__ Uc,
                              const float* __restrict__ h0,
                              float* __restrict__ Hs) {
  const int c = blockIdx.x * 256 + threadIdx.x;
  if (c >= BP) return;
  float h = h0[c];                       // hidden_state [B,P] flat == channel
  for (int j = 0; j < NCHUNK; ++j) {
    Hs[(size_t)j * BP + c] = h;
    h = Fc[(size_t)j * BP + c] * h + Uc[(size_t)j * BP + c];
  }
}

__global__ void k_scan_apply(const float* __restrict__ G,
                             const float* __restrict__ Hs,
                             __bf16* __restrict__ Obf) {        // [M, KPAD]
  const int c = blockIdx.x * 256 + threadIdx.x;
  if (c >= BP) return;
  const int j = blockIdx.y;
  const int b = c / PP, p = c % PP;
  float h = Hs[(size_t)j * BP + c];
  for (int t = 0; t < LCH; ++t) {
    const size_t m = (size_t)b * SS + (size_t)j * LCH + t;
    const float* row = G + m * NSTRIDE;
    const float gi = __builtin_nontemporal_load(row + p);
    const float gf = __builtin_nontemporal_load(row + PP + p);
    const float go = __builtin_nontemporal_load(row + 2 * PP + p);
    const float gc = __builtin_nontemporal_load(row + 3 * PP + p);
    h = softcap_sig(gf) * h + softcap_sig(gi) * tanhf(gc);
    Obf[m * KPAD + p] = (__bf16)(softcap_sig(go) * tanhf(h));
  }
}

__global__ void k_pad_outputs(__bf16* __restrict__ Obf) {
  const int i = blockIdx.x * 256 + threadIdx.x;
  const int m = i / (KPAD - PP), k = PP + i % (KPAD - PP);
  if (m < MM) Obf[(size_t)m * KPAD + k] = (__bf16)0.f;
}

// ---------------------------------------------------------------------------
// 5) Output GEMM + residual: out = x + outputs @ W_out
//    Wave tile 32x64, grid = (16, MM/256), guard-free, EXEC all-ones.
// ---------------------------------------------------------------------------
__global__ void k_gemm_out(const __bf16* __restrict__ O,    // [M, KPAD]
                           const __bf16* __restrict__ WoT,  // [D, KPAD]
                           const float* __restrict__ X,
                           float* __restrict__ Out) {
  const int lane = threadIdx.x & 31;
  const int wave = threadIdx.x >> 5;
  const int m0   = (blockIdx.y * 8 + wave) * 32;
  const int n0   = blockIdx.x * 64;
  f32x8 acc0[4] = {}, acc1[4] = {};
  for (int kk = 0; kk < KPAD; kk += 32) {
    const bf16x16 a0 = load_a_frag(O, KPAD, m0,      kk, lane);
    const bf16x16 a1 = load_a_frag(O, KPAD, m0 + 16, kk, lane);
#pragma unroll
    for (int t = 0; t < 4; ++t) {
      const bf16x16 b = load_b_frag(WoT, KPAD, n0 + t * 16, kk, lane);
      acc0[t] = __builtin_amdgcn_wmma_f32_16x16x32_bf16(
          false, a0, false, b, (short)0, acc0[t], false, false);
      acc1[t] = __builtin_amdgcn_wmma_f32_16x16x32_bf16(
          false, a1, false, b, (short)0, acc1[t], false, false);
    }
  }
  const int mb0 = m0 + (lane >> 4) * 8;
#pragma unroll
  for (int t = 0; t < 4; ++t) {
    const int n = n0 + t * 16 + (lane & 15);
#pragma unroll
    for (int r = 0; r < 8; ++r) {
      const size_t i0 = (size_t)(mb0 + r) * DD + n;
      const size_t i1 = (size_t)(mb0 + 16 + r) * DD + n;
      Out[i0] = X[i0] + acc0[t][r];
      Out[i1] = X[i1] + acc1[t][r];
    }
  }
}

// ---------------------------------------------------------------------------
// 6) Return initial hidden_state (reference returns it unmodified)
// ---------------------------------------------------------------------------
__global__ void k_copy_hidden(const float* __restrict__ h0, float* __restrict__ dst) {
  const int i = blockIdx.x * 256 + threadIdx.x;
  if (i < BP) dst[i] = h0[i];
}

// ---------------------------------------------------------------------------
extern "C" void kernel_launch(void* const* d_in, const int* in_sizes, int n_in,
                              void* d_out, int out_size, void* d_ws, size_t ws_size,
                              hipStream_t stream) {
  const float* x   = (const float*)d_in[0];   // [B,S,D]
  const float* h0  = (const float*)d_in[1];   // [B,P]
  const float* lnw = (const float*)d_in[2];   // [D]
  const float* Wg  = (const float*)d_in[3];   // [D,3P]
  const float* Wc  = (const float*)d_in[4];   // [D,P]
  const float* Wo  = (const float*)d_in[5];   // [P,D]
  float* out = (float*)d_out;

  // workspace carve-up (256B aligned)
  char* ws = (char*)d_ws;
  auto alloc = [&](size_t bytes) {
    char* p = ws;
    ws += (bytes + 255) & ~size_t(255);
    return p;
  };
  __bf16* Xn  = (__bf16*)alloc((size_t)MM * DD * 2);
  __bf16* WT  = (__bf16*)alloc((size_t)NSTRIDE * DD * 2);
  __bf16* WoT = (__bf16*)alloc((size_t)DD * KPAD * 2);
  float*  G   = (float*) alloc((size_t)MM * NSTRIDE * 4);
  __bf16* Obf = (__bf16*)alloc((size_t)MM * KPAD * 2);
  float*  Fc  = (float*) alloc((size_t)NCHUNK * BP * 4);
  float*  Uc  = (float*) alloc((size_t)NCHUNK * BP * 4);
  float*  Hs  = (float*) alloc((size_t)NCHUNK * BP * 4);

  // 1) RMSNorm + bf16 convert
  k_rmsnorm_bf16<<<MM, 256, 0, stream>>>(x, lnw, Xn);

  // 2) weight transposes (bf16)
  k_build_wt_fused<<<dim3(DD / 32, NSTRIDE / 32), dim3(32, 8), 0, stream>>>(Wg, Wc, WT);
  k_build_wout_t<<<dim3(KPAD / 32, DD / 32), dim3(32, 8), 0, stream>>>(Wo, WoT);

  // 3) fused gate+cell GEMM: 86 N-jobs x 64 M-groups (32 rows/wave)
  k_gemm_gates<<<dim3(NJOBS1, MM / 256), 256, 0, stream>>>(Xn, WT, G);

  // 4) chunked scan
  k_scan_chunks<<<dim3((BP + 255) / 256, NCHUNK), 256, 0, stream>>>(G, Fc, Uc);
  k_scan_prefix<<<(BP + 255) / 256, 256, 0, stream>>>(Fc, Uc, h0, Hs);
  k_scan_apply<<<dim3((BP + 255) / 256, NCHUNK), 256, 0, stream>>>(G, Hs, Obf);
  k_pad_outputs<<<(MM * (KPAD - PP) + 255) / 256, 256, 0, stream>>>(Obf);

  // 5) output GEMM + residual: 16 N-jobs x 64 M-groups
  k_gemm_out<<<dim3(16, MM / 256), 256, 0, stream>>>(Obf, WoT, x, out);

  // 6) tail: unmodified hidden_state
  k_copy_hidden<<<(BP + 255) / 256, 256, 0, stream>>>(h0, out + (size_t)MM * DD);
}
